// ESN_40338332844335
// MI455X (gfx1250) — matrix operand order repaired and checked by reading, not compiled, not run on hardware
//
#include <hip/hip_runtime.h>
#include <cmath>

// Problem dims (fixed by reference)
#define B_   64
#define T_   1024
#define IN_  128
#define R_   2048
#define O_   256
#define KC_  (R_ + IN_)          // fused K = 2176
#define NWG  32                  // persistent workgroups, each owns 64 output cols
#define NCOL 64
#define TPB  256                 // 8 waves (wave32)
#define PANEL_K   128            // A staging panel (K chunk)
#define NPANEL    (KC_ / PANEL_K)  // 17
#define PANEL_PITCH 136          // 128 + 8 halves pad (LDS bank spread)
#define PANEL_SZ  (B_ * PANEL_PITCH) // halves per panel buffer
#define KCP  2184                // 2176 + 8 halves pad for B rows in LDS

typedef _Float16 half_t;
typedef __attribute__((ext_vector_type(16))) _Float16 v16h;
typedef __attribute__((ext_vector_type(8)))  float    v8f;

union Frag32B { uint4 q[2]; v16h v; };

// A fragment (16x32 f16): lane holds row m, halves K = {kb+8g+0..7, kb+16+8g+0..7}
__device__ __forceinline__ v16h load_frag_A(const half_t* row, int kb, int g) {
  Frag32B f;
  f.q[0] = *(const uint4*)(row + kb + 8 * g);
  f.q[1] = *(const uint4*)(row + kb + 16 + 8 * g);
  return f.v;
}
// B fragment (32x16 f16): lane holds col n, halves K = {kb+8g+0..7, kb+16+8g+0..7}
__device__ __forceinline__ v16h load_frag_B(const half_t* row, int kb, int g) {
  Frag32B f;
  f.q[0] = *(const uint4*)(row + kb + 8 * g);
  f.q[1] = *(const uint4*)(row + kb + 16 + 8 * g);
  return f.v;
}

// ---- prep: fused f16 weights [R][KC] = [W_res | W_in], fused bias ----
__global__ void esn_prep_weights(const float* __restrict__ Wres,
                                 const float* __restrict__ Win,
                                 const float* __restrict__ bres,
                                 const float* __restrict__ bin,
                                 half_t* __restrict__ Wc, float* __restrict__ bc) {
  const int n = blockIdx.x;                       // reservoir row 0..2047
  for (int k = threadIdx.x; k < KC_; k += TPB) {
    float w = (k < R_) ? Wres[(size_t)n * R_ + k] : Win[(size_t)n * IN_ + (k - R_)];
    Wc[(size_t)n * KC_ + k] = (half_t)w;
  }
  if (threadIdx.x == 0) bc[n] = bres[n] + bin[n];
}

__global__ void esn_prep_input(const float* __restrict__ in, half_t* __restrict__ outp, int n) {
  for (int i = blockIdx.x * blockDim.x + threadIdx.x; i < n; i += gridDim.x * blockDim.x)
    outp[i] = (half_t)in[i];
}

__global__ void esn_init(half_t* __restrict__ h0, unsigned* __restrict__ bar, int n) {
  int i = blockIdx.x * blockDim.x + threadIdx.x;
  if (i == 0) *bar = 0u;
  for (; i < n; i += gridDim.x * blockDim.x) h0[i] = (half_t)0.0f;
}

// Issue global loads for one A panel (registers only; no waits forced here).
__device__ __forceinline__ void load_panel(uint4 v[4], const half_t* __restrict__ hsrc,
                                           const half_t* __restrict__ inF16,
                                           int p, int t, int tid) {
  const int kb0 = p * PANEL_K;
  #pragma unroll
  for (int j = 0; j < 4; ++j) {
    int idx8 = tid + j * TPB;                   // 1024 chunks of 8 halves
    int row  = idx8 >> 4;
    int col  = (idx8 & 15) << 3;
    if (p < NPANEL - 1)
      v[j] = *(const uint4*)(hsrc + (size_t)row * R_ + kb0 + col);
    else
      v[j] = *(const uint4*)(inF16 + ((size_t)row * T_ + t) * IN_ + col);
  }
}

__device__ __forceinline__ void store_panel(half_t* __restrict__ buf, const uint4 v[4], int tid) {
  #pragma unroll
  for (int j = 0; j < 4; ++j) {
    int idx8 = tid + j * TPB;
    int row  = idx8 >> 4;
    int col  = (idx8 & 15) << 3;
    *(uint4*)(buf + row * PANEL_PITCH + col) = v[j];
  }
}

// ---- persistent recurrent kernel: whole T loop, weights resident in LDS ----
__global__ __launch_bounds__(TPB) void esn_recurrent(
    const half_t* __restrict__ inF16,   // [B][T][IN] f16
    const half_t* __restrict__ Wc,      // [R][KC]    f16 fused weights
    const float*  __restrict__ bc,      // [R]        fused bias
    half_t* __restrict__ h0,            // [B][R] f16 state ping
    half_t* __restrict__ h1,            // [B][R] f16 state pong
    float*  __restrict__ hfin,          // [B][R] f32 final state
    unsigned* __restrict__ bar)         // device-wide barrier counter
{
  extern __shared__ char smem[];
  half_t* ldsB = (half_t*)smem;                                        // [NCOL][KCP]
  half_t* ldsA = (half_t*)(smem + (size_t)NCOL * KCP * sizeof(half_t)); // [2][B_][PANEL_PITCH]

  const int tid  = threadIdx.x;
  const int lane = tid & 31;
  const int wave = tid >> 5;
  const int lr   = lane & 15;
  const int g    = lane >> 4;
  const int nt   = wave & 3;            // N-tile 0..3 within the 64-col slice
  const int mh   = wave >> 2;           // M half 0..1 -> M tiles {2mh, 2mh+1}
  const int n0   = blockIdx.x * NCOL;
  const int ng   = n0 + nt * 16 + lr;   // this lane's reservoir column in C/D

  // Load this WG's 64x2176 f16 weight slice into LDS once (reused for all 1024 steps).
  {
    const uint4* src = (const uint4*)(Wc + (size_t)n0 * KC_);
    const int rowChunks = KC_ / 8;                  // 272 x 16B per row
    for (int i = tid; i < NCOL * rowChunks; i += TPB) {
      int r = i / rowChunks, c = i - r * rowChunks;
      *(uint4*)(ldsB + (size_t)r * KCP + c * 8) = src[(size_t)r * rowChunks + c];
    }
  }
  __syncthreads();

  const half_t* hsrc = h0;
  half_t* hdst = h1;
  const float bias = bc[ng];
  const half_t* aRow0 = ldsA + (32 * mh + lr) * PANEL_PITCH;        // M tile 2mh
  const half_t* aRow1 = ldsA + (32 * mh + 16 + lr) * PANEL_PITCH;   // M tile 2mh+1
  const half_t* bRow  = ldsB + (size_t)(nt * 16 + lr) * KCP;

  for (int t = 0; t < T_; ++t) {
    v8f acc0 = {};
    v8f acc1 = {};
    uint4 stage[4];

    // Prime panel 0 (previous users of buf0 are fenced by the grid-barrier syncthreads).
    load_panel(stage, hsrc, inF16, 0, t, tid);
    store_panel(ldsA, stage, tid);
    __syncthreads();

    for (int p = 0; p < NPANEL; ++p) {
      const int cur = (p & 1) * PANEL_SZ;
      // Kick off next panel's global loads before consuming this one (latency hiding).
      if (p + 1 < NPANEL) load_panel(stage, hsrc, inF16, p + 1, t, tid);

      const int kb0 = p * PANEL_K;
      #pragma unroll
      for (int kk = 0; kk < PANEL_K / 32; ++kk) {
        const int kb = kk * 32;
        v16h a0 = load_frag_A(aRow0 + cur, kb, g);
        v16h a1 = load_frag_A(aRow1 + cur, kb, g);
        v16h b  = load_frag_B(bRow, kb0 + kb, g);
        acc0 = __builtin_amdgcn_wmma_f32_16x16x32_f16(false, a0, false, b, (short)0, acc0, false, false);
        acc1 = __builtin_amdgcn_wmma_f32_16x16x32_f16(false, a1, false, b, (short)0, acc1, false, false);
      }

      if (p + 1 < NPANEL) {
        store_panel(ldsA + ((p + 1) & 1) * PANEL_SZ, stage, tid);
        __syncthreads();   // publishes panel p+1; also fences reuse of buf[(p+1)&1]
      }
    }

    // Epilogue: h_new = tanh(acc + bias). C/D layout: VGPR r -> M = mbase + 8g + r.
    const int mb0 = 32 * mh + 8 * g;
    const int mb1 = mb0 + 16;
    #pragma unroll
    for (int r = 0; r < 8; ++r) {
      float v0 = tanhf(acc0[r] + bias);
      float v1 = tanhf(acc1[r] + bias);
      hdst[(size_t)(mb0 + r) * R_ + ng] = (half_t)v0;
      hdst[(size_t)(mb1 + r) * R_ + ng] = (half_t)v1;
      if (t == T_ - 1) {
        hfin[(size_t)(mb0 + r) * R_ + ng] = v0;
        hfin[(size_t)(mb1 + r) * R_ + ng] = v1;
      }
    }
    { half_t* tmp = hdst; hdst = (half_t*)hsrc; hsrc = tmp; }

    // Device-wide step barrier (monotonic counter, agent-scope release/acquire).
    if (t + 1 < T_) {
      __syncthreads();
      if (tid == 0) {
        __hip_atomic_fetch_add(bar, 1u, __ATOMIC_ACQ_REL, __HIP_MEMORY_SCOPE_AGENT);
        const unsigned target = (unsigned)NWG * (unsigned)(t + 1);
        while (__hip_atomic_load(bar, __ATOMIC_ACQUIRE, __HIP_MEMORY_SCOPE_AGENT) < target)
          __builtin_amdgcn_s_sleep(1);
      }
      __syncthreads();
      (void)__hip_atomic_load(bar, __ATOMIC_ACQUIRE, __HIP_MEMORY_SCOPE_AGENT);
    }
  }
}

// ---- readout: out[64,256] = h_final @ W_out^T + b (f32, tiny) ----
__global__ void esn_head(const float* __restrict__ hfin, const float* __restrict__ Ww,
                         const float* __restrict__ Wb, float* __restrict__ out) {
  const int m = blockIdx.x;        // 0..63
  const int o = threadIdx.x;       // 0..255
  const float* h = hfin + (size_t)m * R_;
  const float* w = Ww + (size_t)o * R_;
  float s = Wb[o];
  #pragma unroll 8
  for (int k = 0; k < R_; ++k) s = fmaf(h[k], w[k], s);
  out[m * O_ + o] = s;
}

extern "C" void kernel_launch(void* const* d_in, const int* in_sizes, int n_in,
                              void* d_out, int out_size, void* d_ws, size_t ws_size,
                              hipStream_t stream) {
  const float* input = (const float*)d_in[0];
  const float* Winw  = (const float*)d_in[1];
  const float* Winb  = (const float*)d_in[2];
  const float* Wresw = (const float*)d_in[3];
  const float* Wresb = (const float*)d_in[4];
  const float* Woutw = (const float*)d_in[5];
  const float* Woutb = (const float*)d_in[6];
  float* out = (float*)d_out;

  char* ws = (char*)d_ws;
  size_t off = 0;
  auto wsalloc = [&](size_t bytes) -> void* {
    void* p = ws + off;
    off = (off + bytes + 255) & ~(size_t)255;
    return p;
  };
  half_t*   Wc   = (half_t*)wsalloc(sizeof(half_t) * (size_t)R_ * KC_);      // 8.9 MB
  half_t*   inH  = (half_t*)wsalloc(sizeof(half_t) * (size_t)B_ * T_ * IN_); // 16 MB
  float*    bc   = (float*) wsalloc(sizeof(float) * R_);
  half_t*   h0   = (half_t*)wsalloc(sizeof(half_t) * (size_t)B_ * R_);
  half_t*   h1   = (half_t*)wsalloc(sizeof(half_t) * (size_t)B_ * R_);
  float*    hfin = (float*) wsalloc(sizeof(float) * (size_t)B_ * R_);
  unsigned* bar  = (unsigned*)wsalloc(256);
  (void)ws_size; (void)in_sizes; (void)n_in; (void)out_size;

  const size_t smem = (size_t)NCOL * KCP * sizeof(half_t)            // 279,552 B weights
                    + 2 * (size_t)PANEL_SZ * sizeof(half_t);         //  34,816 B A panels
  (void)hipFuncSetAttribute(reinterpret_cast<const void*>(&esn_recurrent),
                            hipFuncAttributeMaxDynamicSharedMemorySize, (int)smem);

  esn_prep_weights<<<R_, TPB, 0, stream>>>(Wresw, Winw, Wresb, Winb, Wc, bc);
  esn_prep_input <<<2048, TPB, 0, stream>>>(input, inH, B_ * T_ * IN_);
  esn_init       <<<512, TPB, 0, stream>>>(h0, bar, B_ * R_);
  esn_recurrent  <<<NWG, TPB, smem, stream>>>(inH, Wc, bc, h0, h1, hfin, bar);
  esn_head       <<<B_, O_, 0, stream>>>(hfin, Woutw, Woutb, out);
}
